// IpaBlock_65635690217859
// MI455X (gfx1250) — matrix operand order repaired and checked by reading, not compiled, not run on hardware
//
#include <hip/hip_runtime.h>
#include <hip/hip_bf16.h>
#include <math.h>

// ---------------------------------------------------------------------------
// IPA-style graph attention block for MI455X (gfx1250, wave32, WMMA).
//
// Pipeline:
//   1. gemm_wmma_bf16 : h @ [Wq|Wk|Wv|W_fd1|W_g1]^T (+bias, SiLU on fd1/g1)
//                       -> fused QKVTG buffer (N x 1280), bf16 WMMA, f32 acc
//   2. node_small     : sixd = T@W_fd2^T, gate = tanh(G@W_g2^T),
//                       rot-from-6d, base_frames = frames @ dR
//   3. edge_logits    : per-edge q.k per head + all scalar/geometric biases,
//                       atomicMax (monotone-uint) segment max
//   4. edge_ev        : ev = exp(l - m[src]), atomicAdd segment denominator
//   5. edge_agg       : w = ev/den, scatter w*v[dst] into h_upd,
//                       wm-weighted d_loc / d_glb
//   6. gemm_wmma_bf16 : h_out = h_upd @ Wo^T + bo   -> d_out[0 : N*256)
//   7. x_out          : x + gate * 0.5*(frames@d_loc + d_glb) -> d_out[N*256:)
// ---------------------------------------------------------------------------

typedef __attribute__((ext_vector_type(16))) __bf16 v16bf;
typedef __attribute__((ext_vector_type(8)))  __bf16 v8bf;
typedef __attribute__((ext_vector_type(8)))  float  v8f;

#define HIDc  256
#define Hc    8
#define Dc    32
#define NOUT5 1280   // Q|K|V|T|G fused output width
#define LDT   136    // LDS row stride (bf16) for A: 64 rows x 128-K chunk
#define LDBT  72     // LDS row stride (bf16) for B^T: 128 K-rows x 64 cols

// ---- monotone-uint float encoding: atomicMax on uint == max on float ------
__device__ __forceinline__ unsigned encodeF(float f) {
  unsigned u = __float_as_uint(f);
  return (u & 0x80000000u) ? ~u : (u | 0x80000000u);
}
__device__ __forceinline__ float decodeF(unsigned u) {
  return (u & 0x80000000u) ? __uint_as_float(u ^ 0x80000000u)
                           : __uint_as_float(~u);
}

// ---------------------------------------------------------------------------
// Fused multi-weight GEMM: C[M x Nout] = A[M x 256] @ Wseg^T + bseg
// Column segment s = gcol>>8 selects (W_s, b_s); siluMask bit s applies SiLU.
// 64x64 tile per block, 4 waves, each wave: 16 rows x four 16x16 WMMA tiles.
// A staged row-major (K contiguous); W staged TRANSPOSED (BsT[k][col]) so a
// B fragment is 16 bf16 at stride LDBT -> gathered by 16 scalar LDS reads.
// ---------------------------------------------------------------------------
__global__ __launch_bounds__(128)
void gemm_wmma_bf16(const float* __restrict__ A, int lda,
                    const float* __restrict__ W0, const float* __restrict__ W1,
                    const float* __restrict__ W2, const float* __restrict__ W3,
                    const float* __restrict__ W4,
                    const float* __restrict__ b0, const float* __restrict__ b1,
                    const float* __restrict__ b2, const float* __restrict__ b3,
                    const float* __restrict__ b4,
                    float* __restrict__ C, int Nout, int siluMask)
{
  __shared__ __bf16 As [64  * LDT];   // [row][k]
  __shared__ __bf16 BsT[128 * LDBT];  // [k][col]  (transposed weights)

  const int tid     = threadIdx.x;
  const int rowBase = blockIdx.x * 64;
  const int colBase = blockIdx.y * 64;

  const int wave = tid >> 5;
  const int lane = tid & 31;
  const int mIdx = lane & 15;       // A row in tile / C column in tile
  const int hi   = lane >> 4;       // half-wave selector
  const int rowA = wave * 16 + mIdx;
  const int klB  = mIdx + 16 * hi;  // B fragment K index within 32-chunk

  v8f acc[4];
  #pragma unroll
  for (int n = 0; n < 4; ++n)
    #pragma unroll
    for (int i = 0; i < 8; ++i) acc[n][i] = 0.0f;

  for (int kt = 0; kt < 256; kt += 128) {
    __syncthreads();
    // ---- stage A chunk: 64 rows x 128 K (f32 -> bf16), K-contiguous ----
    {
      int r  = tid >> 1;
      int c0 = (tid & 1) * 64;
      const float4* s4 = (const float4*)(A + (size_t)(rowBase + r) * lda + kt + c0);
      __bf16* dl = As + r * LDT + c0;
      #pragma unroll
      for (int j = 0; j < 16; ++j) {
        float4 f = s4[j];
        dl[4*j+0] = (__bf16)f.x; dl[4*j+1] = (__bf16)f.y;
        dl[4*j+2] = (__bf16)f.z; dl[4*j+3] = (__bf16)f.w;
      }
    }
    // ---- stage B chunk TRANSPOSED: BsT[k][c] = W[colBase+c][kt+k] ----
    {
      int c  = tid >> 1;            // local output column 0..63
      int c0 = (tid & 1) * 64;      // K half of the chunk
      int gcol = colBase + c;
      int seg  = gcol >> 8;
      const float* Ws = (seg == 0) ? W0 : (seg == 1) ? W1 : (seg == 2) ? W2
                        : (seg == 3) ? W3 : W4;
      const float4* s4 = (const float4*)(Ws + (size_t)(gcol & 255) * 256 + kt + c0);
      #pragma unroll
      for (int j = 0; j < 16; ++j) {
        float4 f = s4[j];
        int k = c0 + 4 * j;
        BsT[(k + 0) * LDBT + c] = (__bf16)f.x;
        BsT[(k + 1) * LDBT + c] = (__bf16)f.y;
        BsT[(k + 2) * LDBT + c] = (__bf16)f.z;
        BsT[(k + 3) * LDBT + c] = (__bf16)f.w;
      }
    }
    __syncthreads();

    // ---- 4 WMMA K-steps of 32 over this 128-chunk ----
    #pragma unroll
    for (int k0 = 0; k0 < 128; k0 += 32) {
      // A fragment: lane -> row, VGPR0-3: K = k0+8*hi+{0..7},
      //             VGPR4-7: K = k0+16+8*hi+{0..7}. Two aligned b128 loads.
      const v8bf* ap = (const v8bf*)(As + rowA * LDT + k0 + 8 * hi);
      v8bf alo = ap[0];
      v8bf ahi = ap[2];   // +16 bf16
      v16bf a = __builtin_shufflevector(alo, ahi,
                  0,1,2,3,4,5,6,7, 8,9,10,11,12,13,14,15);

      #pragma unroll
      for (int n = 0; n < 4; ++n) {
        // B fragment: lane -> N=mIdx; element s -> K in same order as A
        // (8*hi+s for s<8, 16+8*hi+s-8 for s>=8), gathered from transposed tile.
        v16bf b;
        #pragma unroll
        for (int s = 0; s < 16; ++s) {
          int kk = (s < 8) ? (8 * hi + s) : (16 + 8 * hi + (s - 8));
          b[s] = BsT[(size_t)(k0 + kk) * LDBT + n * 16 + mIdx];
        }
        acc[n] = __builtin_amdgcn_wmma_f32_16x16x32_bf16(
            false, a, false, b, (short)0, acc[n], false, false);
      }
    }
  }

  // ---- epilogue: C layout lane->(N=mIdx), VGPR r -> M = 8*hi + r ----
  #pragma unroll
  for (int n = 0; n < 4; ++n) {
    int gcol = colBase + n * 16 + mIdx;
    int seg  = gcol >> 8;
    const float* bs = (seg == 0) ? b0 : (seg == 1) ? b1 : (seg == 2) ? b2
                      : (seg == 3) ? b3 : b4;
    float bias   = bs[gcol & 255];
    bool  doSilu = (siluMask >> seg) & 1;
    #pragma unroll
    for (int r = 0; r < 8; ++r) {
      int   grow = rowBase + wave * 16 + hi * 8 + r;
      float v    = acc[n][r] + bias;
      if (doSilu) v = v / (1.0f + __expf(-v));
      C[(size_t)grow * Nout + gcol] = v;
    }
  }
}

// ---------------------------------------------------------------------------
// Per-node: sixd = T@W_fd2^T + b, gate = tanh(G@W_g2^T + b),
//           base_frames = frames @ rot_from_6d(sixd)
// ---------------------------------------------------------------------------
__global__ void node_small_kernel(const float* __restrict__ qkvtg,
                                  const float* __restrict__ W_fd2,
                                  const float* __restrict__ b_fd2,
                                  const float* __restrict__ W_g2,
                                  const float* __restrict__ b_g2,
                                  const float* __restrict__ frames,
                                  float* __restrict__ bframes,
                                  float* __restrict__ gatev, int Nn)
{
  int n = blockIdx.x * blockDim.x + threadIdx.x;
  if (n >= Nn) return;
  const float* T = qkvtg + (size_t)n * NOUT5 + 3 * HIDc;  // silu(fd1)
  const float* G = qkvtg + (size_t)n * NOUT5 + 4 * HIDc;  // silu(g1)

  float s[6];
  #pragma unroll
  for (int o = 0; o < 6; ++o) s[o] = b_fd2[o];
  float g = b_g2[0];
  for (int k = 0; k < HIDc; ++k) {
    float tv = T[k];
    #pragma unroll
    for (int o = 0; o < 6; ++o) s[o] += tv * W_fd2[o * HIDc + k];
    g += G[k] * W_g2[k];
  }
  g = tanhf(g);

  float a1[3] = {s[0], s[1], s[2]}, a2[3] = {s[3], s[4], s[5]};
  float n1 = sqrtf(a1[0]*a1[0] + a1[1]*a1[1] + a1[2]*a1[2]);
  float i1 = 1.0f / fmaxf(n1, 1e-6f);
  float b1[3] = {a1[0]*i1, a1[1]*i1, a1[2]*i1};
  float dp = b1[0]*a2[0] + b1[1]*a2[1] + b1[2]*a2[2];
  float c2[3] = {a2[0]-dp*b1[0], a2[1]-dp*b1[1], a2[2]-dp*b1[2]};
  float n2 = sqrtf(c2[0]*c2[0] + c2[1]*c2[1] + c2[2]*c2[2]);
  float i2 = 1.0f / fmaxf(n2, 1e-6f);
  float b2[3] = {c2[0]*i2, c2[1]*i2, c2[2]*i2};
  float b3[3] = {b1[1]*b2[2]-b1[2]*b2[1],
                 b1[2]*b2[0]-b1[0]*b2[2],
                 b1[0]*b2[1]-b1[1]*b2[0]};
  float R[3][3];
  #pragma unroll
  for (int i = 0; i < 3; ++i) { R[i][0]=b1[i]; R[i][1]=b2[i]; R[i][2]=b3[i]; }

  const float* F = frames + (size_t)n * 9;
  #pragma unroll
  for (int i = 0; i < 3; ++i)
    #pragma unroll
    for (int j = 0; j < 3; ++j)
      bframes[(size_t)n*9 + i*3 + j] =
          F[i*3+0]*R[0][j] + F[i*3+1]*R[1][j] + F[i*3+2]*R[2][j];
  gatev[n] = g;
}

// ---------------------------------------------------------------------------
// Per-edge logits (all 8 heads) + segment max via encoded atomicMax.
// ---------------------------------------------------------------------------
__global__ __launch_bounds__(128)
void edge_logits_kernel(const int* __restrict__ src, const int* __restrict__ dst,
                        const float* __restrict__ qkvtg,
                        const float* __restrict__ x,
                        const float* __restrict__ frames,
                        const float* __restrict__ bframes,
                        const float* __restrict__ distances,
                        const float* __restrict__ bpp,  const float* __restrict__ msa,
                        const float* __restrict__ chem, const float* __restrict__ rel,
                        const float* __restrict__ chain,
                        const float* __restrict__ w_dist, const float* __restrict__ b_dist,
                        const float* __restrict__ w_bpp,  const float* __restrict__ b_bpp,
                        const float* __restrict__ w_msa,  const float* __restrict__ b_msa,
                        const float* __restrict__ w_chem, const float* __restrict__ b_chem,
                        const float* __restrict__ w_rel,  const float* __restrict__ b_rel,
                        const float* __restrict__ w_chain,const float* __restrict__ b_chain,
                        const float* __restrict__ W_or,   const float* __restrict__ b_or,
                        const float* __restrict__ W_bor,  const float* __restrict__ b_bor,
                        float* __restrict__ logits, float* __restrict__ eldv,
                        unsigned* __restrict__ mmax, int Ee)
{
  int e = blockIdx.x * blockDim.x + threadIdx.x;
  if (e >= Ee) return;
  int s = src[e], d = dst[e];

  const float4* qr = (const float4*)(qkvtg + (size_t)s * NOUT5);            // Q row
  const float4* kr = (const float4*)(qkvtg + (size_t)d * NOUT5 + HIDc);     // K row
  float lg[Hc];
  #pragma unroll
  for (int h = 0; h < Hc; ++h) {
    float acc = 0.0f;
    #pragma unroll
    for (int j = 0; j < 8; ++j) {
      float4 a = qr[h*8 + j], b = kr[h*8 + j];
      acc += a.x*b.x + a.y*b.y + a.z*b.z + a.w*b.w;
    }
    lg[h] = acc * 0.17677669529663687f;  // 1/sqrt(32)
  }

  float del[3];
  #pragma unroll
  for (int i = 0; i < 3; ++i) del[i] = x[d*3+i] - x[s*3+i];
  float eld[3], bld[3];
  #pragma unroll
  for (int i = 0; i < 3; ++i) {
    eld[i] = frames[(size_t)s*9 + 0*3 + i]*del[0] + frames[(size_t)s*9 + 1*3 + i]*del[1]
           + frames[(size_t)s*9 + 2*3 + i]*del[2];
    bld[i] = bframes[(size_t)s*9 + 0*3 + i]*del[0] + bframes[(size_t)s*9 + 1*3 + i]*del[1]
           + bframes[(size_t)s*9 + 2*3 + i]*del[2];
    eldv[(size_t)e*3 + i] = eld[i];
  }
  float en = sqrtf(eld[0]*eld[0] + eld[1]*eld[1] + eld[2]*eld[2]);
  float bn = sqrtf(bld[0]*bld[0] + bld[1]*bld[1] + bld[2]*bld[2]);
  float ei = 1.0f / fmaxf(en, 1e-6f), bi = 1.0f / fmaxf(bn, 1e-6f);
  float eu[3] = {eld[0]*ei, eld[1]*ei, eld[2]*ei};
  float bu[3] = {bld[0]*bi, bld[1]*bi, bld[2]*bi};

  float d2 = distances[e]; d2 *= d2;
  float bp = bpp[e], ms = msa[e], ch = chem[e], ro = rel[e], cb = chain[e];

  #pragma unroll
  for (int h = 0; h < Hc; ++h) {
    float l = lg[h]
      - (d2 * w_dist[h] + b_dist[h])
      + (bp * w_bpp[h]  + b_bpp[h])
      + (ms * w_msa[h]  + b_msa[h])
      + (ch * w_chem[h] + b_chem[h])
      + (ro * w_rel[h]  + b_rel[h])
      + (cb * w_chain[h]+ b_chain[h])
      + 0.1f * tanhf(eu[0]*W_or[h*3] + eu[1]*W_or[h*3+1] + eu[2]*W_or[h*3+2] + b_or[h])
      + 0.1f * tanhf(bu[0]*W_bor[h*3]+ bu[1]*W_bor[h*3+1]+ bu[2]*W_bor[h*3+2]+ b_bor[h]);
    logits[(size_t)e*Hc + h] = l;
    atomicMax(mmax + (size_t)s*Hc + h, encodeF(l));
  }
}

// ---------------------------------------------------------------------------
// ev = exp(l - m[src]); den[src] += ev   (logits buffer overwritten with ev)
// ---------------------------------------------------------------------------
__global__ void edge_ev_kernel(const int* __restrict__ src,
                               const unsigned* __restrict__ mmax,
                               float* __restrict__ logits,
                               float* __restrict__ den, int EH)
{
  int idx = blockIdx.x * blockDim.x + threadIdx.x;
  if (idx >= EH) return;
  int e = idx >> 3, h = idx & 7;
  int s = src[e];
  float mm = decodeF(mmax[(size_t)s*Hc + h]);
  float ev = __expf(logits[idx] - mm);
  logits[idx] = ev;
  atomicAdd(den + (size_t)s*Hc + h, ev);
}

// ---------------------------------------------------------------------------
// w = ev/den; h_upd[src] += w * v[dst]; wm-weighted d_loc / d_glb.
// ---------------------------------------------------------------------------
__global__ void edge_agg_kernel(const int* __restrict__ src, const int* __restrict__ dst,
                                const float* __restrict__ qkvtg,
                                const float* __restrict__ evbuf,
                                const float* __restrict__ den,
                                const float* __restrict__ x,
                                const float* __restrict__ eldv,
                                float* __restrict__ hupd,
                                float* __restrict__ dloc, float* __restrict__ dglb,
                                int EH)
{
  int idx = blockIdx.x * blockDim.x + threadIdx.x;
  if (idx >= EH) return;
  int e = idx >> 3, hh = idx & 7;
  int s = src[e], d = dst[e];
  float w = evbuf[idx] / fmaxf(den[(size_t)s*Hc + hh], 1e-9f);
  const float* vrow = qkvtg + (size_t)d * NOUT5 + 2 * HIDc + hh * Dc;
  float* orow = hupd + (size_t)s * HIDc + hh * Dc;
  #pragma unroll
  for (int i = 0; i < Dc; ++i) atomicAdd(orow + i, w * vrow[i]);

  if (hh == 0) {
    float wm = 0.0f;
    #pragma unroll
    for (int h2 = 0; h2 < Hc; ++h2)
      wm += evbuf[(size_t)e*Hc + h2] / fmaxf(den[(size_t)s*Hc + h2], 1e-9f);
    wm *= 0.125f;
    #pragma unroll
    for (int i = 0; i < 3; ++i) {
      atomicAdd(dglb + (size_t)s*3 + i, wm * (x[d*3+i] - x[s*3+i]));
      atomicAdd(dloc + (size_t)s*3 + i, wm * eldv[(size_t)e*3 + i]);
    }
  }
}

// ---------------------------------------------------------------------------
// x_out = x + gate * 0.5*(frames @ d_loc + d_glb)
// ---------------------------------------------------------------------------
__global__ void x_out_kernel(const float* __restrict__ x,
                             const float* __restrict__ frames,
                             const float* __restrict__ dloc,
                             const float* __restrict__ dglb,
                             const float* __restrict__ gatev,
                             float* __restrict__ xout, int Nn)
{
  int n = blockIdx.x * blockDim.x + threadIdx.x;
  if (n >= Nn) return;
  float g = gatev[n];
  const float* F = frames + (size_t)n * 9;
  #pragma unroll
  for (int i = 0; i < 3; ++i) {
    float disp = 0.5f * (F[i*3+0]*dloc[(size_t)n*3+0] + F[i*3+1]*dloc[(size_t)n*3+1]
                       + F[i*3+2]*dloc[(size_t)n*3+2] + dglb[(size_t)n*3+i]);
    xout[(size_t)n*3 + i] = x[(size_t)n*3 + i] + g * disp;
  }
}

// ---------------------------------------------------------------------------
extern "C" void kernel_launch(void* const* d_in, const int* in_sizes, int n_in,
                              void* d_out, int out_size, void* d_ws, size_t ws_size,
                              hipStream_t stream)
{
  // ---- inputs, in setup_inputs() dict order ----
  const float* h         = (const float*)d_in[0];
  const float* x         = (const float*)d_in[1];
  const int*   src       = (const int*)  d_in[2];
  const int*   dst       = (const int*)  d_in[3];
  const float* distances = (const float*)d_in[4];
  const float* bpp       = (const float*)d_in[5];
  const float* msa       = (const float*)d_in[6];
  const float* chem      = (const float*)d_in[7];
  const float* rel       = (const float*)d_in[8];
  const float* chain     = (const float*)d_in[9];
  const float* frames    = (const float*)d_in[10];
  // params, dict order
  const float* Wq    = (const float*)d_in[11]; const float* bq    = (const float*)d_in[12];
  const float* Wk    = (const float*)d_in[13]; const float* bk    = (const float*)d_in[14];
  const float* Wv    = (const float*)d_in[15]; const float* bv    = (const float*)d_in[16];
  const float* Wo    = (const float*)d_in[17]; const float* bo    = (const float*)d_in[18];
  const float* w_dist= (const float*)d_in[19]; const float* b_dist= (const float*)d_in[20];
  const float* w_bpp = (const float*)d_in[21]; const float* b_bpp = (const float*)d_in[22];
  const float* w_msa = (const float*)d_in[23]; const float* b_msa = (const float*)d_in[24];
  const float* w_chem= (const float*)d_in[25]; const float* b_chem= (const float*)d_in[26];
  const float* w_rel = (const float*)d_in[27]; const float* b_rel = (const float*)d_in[28];
  const float* w_chn = (const float*)d_in[29]; const float* b_chn = (const float*)d_in[30];
  const float* W_or  = (const float*)d_in[31]; const float* b_or  = (const float*)d_in[32];
  const float* W_bor = (const float*)d_in[33]; const float* b_bor = (const float*)d_in[34];
  const float* W_fd1 = (const float*)d_in[35]; const float* b_fd1 = (const float*)d_in[36];
  const float* W_fd2 = (const float*)d_in[37]; const float* b_fd2 = (const float*)d_in[38];
  const float* W_g1  = (const float*)d_in[39]; const float* b_g1  = (const float*)d_in[40];
  const float* W_g2  = (const float*)d_in[41]; const float* b_g2  = (const float*)d_in[42];

  const int Nn = in_sizes[0] / HIDc;   // 8192
  const int Ee = in_sizes[2];          // 131072

  // ---- workspace layout (floats) ----
  float* ws = (float*)d_ws;
  size_t o = 0;
  float*    qkvtg   = ws + o; o += (size_t)Nn * NOUT5;  // Q|K|V|silu(T)|silu(G)
  float*    bframes = ws + o; o += (size_t)Nn * 9;
  float*    gatev   = ws + o; o += (size_t)Nn;
  float*    logits  = ws + o; o += (size_t)Ee * Hc;     // later holds ev
  float*    eldv    = ws + o; o += (size_t)Ee * 3;
  // accumulators (contiguous -> single memset)
  unsigned* mmax    = (unsigned*)(ws + o); o += (size_t)Nn * Hc;
  float*    den     = ws + o; o += (size_t)Nn * Hc;
  float*    hupd    = ws + o; o += (size_t)Nn * HIDc;
  float*    dloc    = ws + o; o += (size_t)Nn * 3;
  float*    dglb    = ws + o; o += (size_t)Nn * 3;

  const size_t accumBytes = (size_t)Nn * (Hc + Hc + HIDc + 3 + 3) * sizeof(float);
  hipMemsetAsync(mmax, 0, accumBytes, stream);  // encoded 0 < encode(-inf): safe floor

  // 1. fused QKV + fd1 + g1 GEMM (SiLU on segments 3,4)
  gemm_wmma_bf16<<<dim3(Nn / 64, NOUT5 / 64), 128, 0, stream>>>(
      h, HIDc, Wq, Wk, Wv, W_fd1, W_g1, bq, bk, bv, b_fd1, b_g1,
      qkvtg, NOUT5, (1 << 3) | (1 << 4));

  // 2. per-node rot / gate
  node_small_kernel<<<(Nn + 255) / 256, 256, 0, stream>>>(
      qkvtg, W_fd2, b_fd2, W_g2, b_g2, frames, bframes, gatev, Nn);

  // 3. edge logits + segment max
  edge_logits_kernel<<<(Ee + 127) / 128, 128, 0, stream>>>(
      src, dst, qkvtg, x, frames, bframes, distances, bpp, msa, chem, rel, chain,
      w_dist, b_dist, w_bpp, b_bpp, w_msa, b_msa, w_chem, b_chem,
      w_rel, b_rel, w_chn, b_chn, W_or, b_or, W_bor, b_bor,
      logits, eldv, mmax, Ee);

  // 4. exp + segment denominator
  edge_ev_kernel<<<(Ee * Hc + 255) / 256, 256, 0, stream>>>(
      src, mmax, logits, den, Ee * Hc);

  // 5. weighted scatter-aggregation
  edge_agg_kernel<<<(Ee * Hc + 255) / 256, 256, 0, stream>>>(
      src, dst, qkvtg, logits, den, x, eldv, hupd, dloc, dglb, Ee * Hc);

  // 6. output projection: h_out -> d_out[0 : N*256)
  gemm_wmma_bf16<<<dim3(Nn / 64, HIDc / 64), 128, 0, stream>>>(
      hupd, HIDc, Wo, Wo, Wo, Wo, Wo, bo, bo, bo, bo, bo,
      (float*)d_out, HIDc, 0);

  // 7. coordinate update -> d_out[N*256 : N*256 + N*3)
  x_out_kernel<<<(Nn + 255) / 256, 256, 0, stream>>>(
      x, frames, dloc, dglb, gatev, (float*)d_out + (size_t)Nn * HIDc, Nn);
}
